// SimpleNetWSage_2542620639565
// MI455X (gfx1250) — matrix operand order repaired and checked by reading, not compiled, not run on hardware
//
#include <hip/hip_runtime.h>

typedef __attribute__((ext_vector_type(16))) _Float16 v16h;
typedef __attribute__((ext_vector_type(8)))  float    v8f;
typedef __attribute__((ext_vector_type(2)))  float    v2f;

#if __has_builtin(__builtin_amdgcn_wmma_f32_16x16x4_f32)
#define USE_F32_WMMA 1
#else
#define USE_F32_WMMA 0
#endif

#define KMAX        256          // max K = 2*Fin (layer 1)
#define LDS_STRIDE  (KMAX + 4)   // floats; stride%64==4 -> bank-conflict-free B reads

// ---------------------------------------------------------------------------
// Zero fill
// ---------------------------------------------------------------------------
__global__ void zero_f32_kernel(float* __restrict__ p, long long n) {
    long long i = (long long)blockIdx.x * blockDim.x + threadIdx.x;
    long long stride = (long long)gridDim.x * blockDim.x;
    for (; i < n; i += stride) p[i] = 0.0f;
}

// ---------------------------------------------------------------------------
// Degree count: cnt[src[e]] += 1
// ---------------------------------------------------------------------------
__global__ void degree_kernel(const long long* __restrict__ src,
                              float* __restrict__ cnt, int E) {
    int e = blockIdx.x * blockDim.x + threadIdx.x;
    if (e < E) atomicAdd(&cnt[(int)src[e]], 1.0f);
}

// ---------------------------------------------------------------------------
// Scatter-add aggregation: agg[src[e], :] += x[dst[e], :]
// One thread handles 4 consecutive features of one edge. For Fin=128 a wave
// covers exactly one edge -> coalesced float4 gather + coalesced atomics.
// ---------------------------------------------------------------------------
__global__ void aggregate_kernel(const long long* __restrict__ src,
                                 const long long* __restrict__ dst,
                                 const float* __restrict__ x,
                                 float* __restrict__ agg,
                                 int E, int Fin) {
    const int chunks = Fin >> 2;
    long long idx = (long long)blockIdx.x * blockDim.x + threadIdx.x;
    long long total = (long long)E * chunks;
    if (idx >= total) return;
    int e = (int)(idx / chunks);
    int c = (int)(idx % chunks);
    int s = (int)src[e];
    int d = (int)dst[e];
    const float4 v = *(const float4*)(x + (size_t)d * Fin + (size_t)c * 4);
    float* o = agg + (size_t)s * Fin + (size_t)c * 4;
    atomicAdd(o + 0, v.x);
    atomicAdd(o + 1, v.y);
    atomicAdd(o + 2, v.z);
    atomicAdd(o + 3, v.w);
}

// ---------------------------------------------------------------------------
// Fused SAGE layer GEMM:
//   out[n,:] = relu( mean[n,:] @ Wl^T + bl + x[n,:] @ Wr^T ),  Fout = 64
// A = [agg/cnt | x] (K = 2*Fin), B = [Wl | Wr]^T staged once in LDS.
// One wave computes a 16-row tile x all 64 output columns (4 WMMA accums).
// ---------------------------------------------------------------------------
__device__ __forceinline__ void load8(const float* __restrict__ base,
                                      float s, float* __restrict__ o) {
    float4 p0 = ((const float4*)base)[0];
    float4 p1 = ((const float4*)base)[1];
    o[0] = p0.x * s; o[1] = p0.y * s; o[2] = p0.z * s; o[3] = p0.w * s;
    o[4] = p1.x * s; o[5] = p1.y * s; o[6] = p1.z * s; o[7] = p1.w * s;
}

__global__ void sage_layer_kernel(const float* __restrict__ agg,
                                  const float* __restrict__ cnt,
                                  const float* __restrict__ xin,
                                  const float* __restrict__ Wl,
                                  const float* __restrict__ Wr,
                                  const float* __restrict__ bias,
                                  float* __restrict__ out,
                                  int nNodes, int Fin) {
    __shared__ float wlds[64 * LDS_STRIDE];           // [Wl|Wr] as f32, padded rows

    const int K = Fin * 2;

    // Cooperative stage of concatenated weights into LDS (once per block).
    for (int idx = threadIdx.x; idx < 64 * K; idx += blockDim.x) {
        const int fo = idx / K;
        const int k  = idx - fo * K;
        wlds[fo * LDS_STRIDE + k] =
            (k < Fin) ? Wl[(size_t)fo * Fin + k]
                      : Wr[(size_t)fo * Fin + (k - Fin)];
    }
    __syncthreads();

    const int lane = threadIdx.x & 31;
    const int wave = threadIdx.x >> 5;
    const int l15  = lane & 15;
    const int half = lane >> 4;                       // wave32 half-select
    const int tileRow = (blockIdx.x * (blockDim.x >> 5) + wave) * 16;
    if (tileRow >= nNodes) return;

    const int  row   = tileRow + l15;                 // A row loaded by this lane
    const bool rowOK = row < nNodes;
    const int  rowc  = rowOK ? row : (nNodes - 1);    // clamped: loads stay in-bounds
    // Invalid lanes are zeroed through the A scale factor (branchless).
    const float invc = 1.0f / fmaxf(cnt[rowc], 1.0f);
    const float sAgg = rowOK ? invc : 0.0f;           // also fuses the mean division
    const float sX   = rowOK ? 1.0f : 0.0f;

    v8f acc[4] = {};                                  // 4 x (16x16) output tiles

#if USE_F32_WMMA
    // ---- Exact f32 path: V_WMMA_F32_16X16X4_F32, K-step = 4 ------------------
    // A 16x4: lane holds K = half*2 + {0,1}; B 4x16 symmetric from LDS.
    auto loadA2 = [&](int ks) -> float2 {             // ks must be < K
        return (ks < Fin) ? *(const float2*)(agg + (size_t)rowc * Fin + ks)
                          : *(const float2*)(xin + (size_t)rowc * Fin + (ks - Fin));
    };

    float2 av = loadA2(half * 2);                     // prefetch step 0
    for (int k0 = 0; k0 < K; k0 += 4) {
        // Branchless prefetch of next step (clamped; final prefetch unused).
        const int ksn = k0 + 4 + half * 2;
        const float2 avn = loadA2(ksn < K ? ksn : half * 2);

        const int   ks    = k0 + half * 2;
        const float scale = (ks < Fin) ? sAgg : sX;
        v2f a; a[0] = av.x * scale; a[1] = av.y * scale;

        #pragma unroll
        for (int t = 0; t < 4; ++t) {
            const v2f b = *(const v2f*)&wlds[(t * 16 + l15) * LDS_STRIDE + ks];
            acc[t] = __builtin_amdgcn_wmma_f32_16x16x4_f32(
                false, a, false, b, (short)0, acc[t], false, false);
        }
        av = avn;
    }
#else
    // ---- Fallback: f16 inputs, f32 accumulate, K-step = 32 -------------------
    for (int k0 = 0; k0 < K; k0 += 32) {
        float af[16];
        #pragma unroll
        for (int g = 0; g < 2; ++g) {
            const int ks = k0 + half * 8 + g * 16;
            float* o = af + g * 8;
            if (ks < Fin) load8(agg + (size_t)rowc * Fin + ks, sAgg, o);
            else          load8(xin + (size_t)rowc * Fin + (ks - Fin), sX, o);
        }
        v16h a;
        #pragma unroll
        for (int j = 0; j < 16; ++j) a[j] = (_Float16)af[j];

        const int kb = k0 + half * 8;
        #pragma unroll
        for (int t = 0; t < 4; ++t) {
            const float* wr = &wlds[(t * 16 + l15) * LDS_STRIDE + kb];
            float wf[16];
            load8(wr,     1.0f, wf);
            load8(wr + 16, 1.0f, wf + 8);
            v16h b;
            #pragma unroll
            for (int j = 0; j < 16; ++j) b[j] = (_Float16)wf[j];
            acc[t] = __builtin_amdgcn_wmma_f32_16x16x32_f16(
                false, a, false, b, (short)0, acc[t], false, false);
        }
    }
#endif

    // ---- Store D: m = tileRow + r + 8*half, n = t*16 + l15 -------------------
    if (tileRow + 16 <= nNodes) {
        // Full tile (always the case when nNodes % 16 == 0): unguarded stores.
        #pragma unroll
        for (int t = 0; t < 4; ++t) {
            const int col = t * 16 + l15;
            const float bc = bias[col];
            #pragma unroll
            for (int r = 0; r < 8; ++r) {
                const int m = tileRow + r + 8 * half;
                out[(size_t)m * 64 + col] = fmaxf(acc[t][r] + bc, 0.0f);
            }
        }
    } else {
        #pragma unroll
        for (int t = 0; t < 4; ++t) {
            const int col = t * 16 + l15;
            const float bc = bias[col];
            #pragma unroll
            for (int r = 0; r < 8; ++r) {
                const int m = tileRow + r + 8 * half;
                if (m < nNodes) {
                    out[(size_t)m * 64 + col] = fmaxf(acc[t][r] + bc, 0.0f);
                }
            }
        }
    }
}

// ---------------------------------------------------------------------------
// Output heads: probs = mean_h @ cpWl^T + cpbl + h @ cpWr^T (Fout=1),
//               dn = h @ Wdn^T + bdn, val = h @ Wv^T + bv  -> out[N,3]
// ---------------------------------------------------------------------------
__global__ void heads_kernel(const float* __restrict__ agg,
                             const float* __restrict__ cnt,
                             const float* __restrict__ h,
                             const float* __restrict__ WlP, const float* __restrict__ blP,
                             const float* __restrict__ WrP,
                             const float* __restrict__ Wdn, const float* __restrict__ bdn,
                             const float* __restrict__ Wv,  const float* __restrict__ bv,
                             float* __restrict__ out, int nNodes) {
    int n = blockIdx.x * blockDim.x + threadIdx.x;
    if (n >= nNodes) return;
    const float invc = 1.0f / fmaxf(cnt[n], 1.0f);
    const float* ha = h   + (size_t)n * 64;
    const float* ag = agg + (size_t)n * 64;
    float p = 0.0f, dn = 0.0f, vv = 0.0f;
    #pragma unroll 8
    for (int k = 0; k < 64; ++k) {
        const float m  = ag[k] * invc;
        const float hh = ha[k];
        p  += m * WlP[k] + hh * WrP[k];
        dn += hh * Wdn[k];
        vv += hh * Wv[k];
    }
    out[(size_t)n * 3 + 0] = p  + blP[0];
    out[(size_t)n * 3 + 1] = dn + bdn[0];
    out[(size_t)n * 3 + 2] = vv + bv[0];
}

// ---------------------------------------------------------------------------
extern "C" void kernel_launch(void* const* d_in, const int* in_sizes, int n_in,
                              void* d_out, int out_size, void* d_ws, size_t ws_size,
                              hipStream_t stream) {
    const int D = 128, H = 64;
    const float*     x  = (const float*)d_in[0];
    const long long* ei = (const long long*)d_in[1];
    const int N = in_sizes[0] / D;
    const int E = in_sizes[1] / 2;
    const long long* src = ei;       // edge_index[0]
    const long long* dst = ei + E;   // edge_index[1]

    const float* c1Wl = (const float*)d_in[2];
    const float* c1bl = (const float*)d_in[3];
    const float* c1Wr = (const float*)d_in[4];
    const float* c2Wl = (const float*)d_in[5];
    const float* c2bl = (const float*)d_in[6];
    const float* c2Wr = (const float*)d_in[7];
    const float* c3Wl = (const float*)d_in[8];
    const float* c3bl = (const float*)d_in[9];
    const float* c3Wr = (const float*)d_in[10];
    const float* c4Wl = (const float*)d_in[11];
    const float* c4bl = (const float*)d_in[12];
    const float* c4Wr = (const float*)d_in[13];
    const float* cpWl = (const float*)d_in[14];
    const float* cpbl = (const float*)d_in[15];
    const float* cpWr = (const float*)d_in[16];
    const float* dnW  = (const float*)d_in[17];
    const float* dnB  = (const float*)d_in[18];
    const float* vW   = (const float*)d_in[19];
    const float* vB   = (const float*)d_in[20];
    (void)n_in; (void)out_size; (void)ws_size;

    // Workspace layout (~51.4 MB total)
    auto align256 = [](size_t v) { return (v + 255) & ~(size_t)255; };
    char* ws = (char*)d_ws;
    size_t off = 0;
    float* cnt = (float*)(ws + off); off += align256((size_t)N * 4);
    float* agg = (float*)(ws + off); off += align256((size_t)N * D * 4);
    float* hA  = (float*)(ws + off); off += align256((size_t)N * H * 4);
    float* hB  = (float*)(ws + off); off += align256((size_t)N * H * 4);

    const int TB = 256;  // 8 waves/block on wave32
    auto cdiv = [](long long a, long long b) { return (int)((a + b - 1) / b); };
    const int gemmGrid = cdiv(cdiv(N, 16), TB / 32);

    // Degree (graph is fixed -> once)
    zero_f32_kernel<<<cdiv(N, TB), TB, 0, stream>>>(cnt, N);
    degree_kernel<<<cdiv(E, TB), TB, 0, stream>>>(src, cnt, E);

    // Layer 1: Fin = 128, x -> hA
    zero_f32_kernel<<<8192, TB, 0, stream>>>(agg, (long long)N * D);
    aggregate_kernel<<<cdiv((long long)E * (D / 4), TB), TB, 0, stream>>>(src, dst, x, agg, E, D);
    sage_layer_kernel<<<gemmGrid, TB, 0, stream>>>(agg, cnt, x, c1Wl, c1Wr, c1bl, hA, N, D);

    // Layer 2: Fin = 64, hA -> hB
    zero_f32_kernel<<<8192, TB, 0, stream>>>(agg, (long long)N * H);
    aggregate_kernel<<<cdiv((long long)E * (H / 4), TB), TB, 0, stream>>>(src, dst, hA, agg, E, H);
    sage_layer_kernel<<<gemmGrid, TB, 0, stream>>>(agg, cnt, hA, c2Wl, c2Wr, c2bl, hB, N, H);

    // Layer 3: hB -> hA
    zero_f32_kernel<<<8192, TB, 0, stream>>>(agg, (long long)N * H);
    aggregate_kernel<<<cdiv((long long)E * (H / 4), TB), TB, 0, stream>>>(src, dst, hB, agg, E, H);
    sage_layer_kernel<<<gemmGrid, TB, 0, stream>>>(agg, cnt, hB, c3Wl, c3Wr, c3bl, hA, N, H);

    // Layer 4: hA -> hB
    zero_f32_kernel<<<8192, TB, 0, stream>>>(agg, (long long)N * H);
    aggregate_kernel<<<cdiv((long long)E * (H / 4), TB), TB, 0, stream>>>(src, dst, hA, agg, E, H);
    sage_layer_kernel<<<gemmGrid, TB, 0, stream>>>(agg, cnt, hA, c4Wl, c4Wr, c4bl, hB, N, H);

    // Final aggregation of hB + heads -> out [N, 3]
    zero_f32_kernel<<<8192, TB, 0, stream>>>(agg, (long long)N * H);
    aggregate_kernel<<<cdiv((long long)E * (H / 4), TB), TB, 0, stream>>>(src, dst, hB, agg, E, H);
    heads_kernel<<<cdiv(N, TB), TB, 0, stream>>>(agg, cnt, hB,
                                                 cpWl, cpbl, cpWr,
                                                 dnW, dnB, vW, vB,
                                                 (float*)d_out, N);
}
